// VisualGraph_15144054686139
// MI455X (gfx1250) — matrix operand order repaired and checked by reading, not compiled, not run on hardware
//
#include <hip/hip_runtime.h>
#include <math.h>

// ---------------- problem constants (from reference) ----------------
#define EMBED  1024
#define KREG   36      // regions per image
#define LW     48      // words per caption (== MAX_WORDS)
#define LWP    64      // words padded for GEMM2 K dim
#define NIMG   64
#define NCAP   32
#define NKER   8
#define NB     32      // NUM_BLOCK
#define BS     32      // N_BLOCK
#define HIDN   256
#define LAM    9.0f
#define EPSF   1e-8f

typedef __attribute__((ext_vector_type(16))) _Float16 v16h;
typedef __attribute__((ext_vector_type(8)))  float    v8f;

// ---- CDNA5 WMMA fragment index helpers (wave32, 16-bit 16x16x32) ----
// A matrix 16x32 (MxK): lane = h*16 + m ; half v (0..15):
//   k = v + h*16   (contiguous, same as B)
__device__ __forceinline__ int akoff(int v, int h) {
    return v + (h << 4);
}
// B matrix 32x16 (KxN): lane = h*16 + n ; half v: k = v + h*16 (contiguous)
__device__ __forceinline__ int bkoff(int v, int h) { return v + (h << 4); }
// C/D 16x16 f32: lane = h*16 + n ; vgpr r: row = r + h*8

__device__ __forceinline__ v8f wmma16(const v16h& a, const v16h& b, const v8f& c) {
    return __builtin_amdgcn_wmma_f32_16x16x32_f16(
        /*neg_a=*/false, a, /*neg_b=*/false, b,
        /*c_mod=*/(short)0, c, /*reuse_a=*/false, /*reuse_b=*/false);
}

// ---------------- prep: gaussian-kernel graph weights ----------------
__global__ void prep_wsum(const float* __restrict__ bbox,
                          const float* __restrict__ mean_rho,
                          const float* __restrict__ mean_theta,
                          const float* __restrict__ prec_rho,
                          const float* __restrict__ prec_theta,
                          float* __restrict__ wsum) {
    int tid = blockIdx.x * blockDim.x + threadIdx.x;
    if (tid >= NIMG * KREG) return;
    int b = tid / KREG, i = tid % KREG;
    const float* bb = bbox + (size_t)b * KREG * 4;
    float cxi = bb[i*4+0] + 0.5f * (bb[i*4+2] - bb[i*4+0]);
    float cyi = bb[i*4+1] + 0.5f * (bb[i*4+3] - bb[i*4+1]);
    float acc[NKER];
    #pragma unroll
    for (int k = 0; k < NKER; ++k) acc[k] = 0.f;
    for (int jj = 0; jj < KREG; ++jj) {
        float cxj = bb[jj*4+0] + 0.5f * (bb[jj*4+2] - bb[jj*4+0]);
        float cyj = bb[jj*4+1] + 0.5f * (bb[jj*4+3] - bb[jj*4+1]);
        float dx = cxi - cxj, dy = cyi - cyj;
        float rho = sqrtf(dx*dx + dy*dy);
        float th  = atan2f(dx, dy);
        #pragma unroll
        for (int k = 0; k < NKER; ++k) {
            float dr = rho - mean_rho[k];
            float wr = __expf(-0.5f * dr * dr / (1e-14f + prec_rho[k]*prec_rho[k]));
            float fa = fabsf(th - mean_theta[k]);
            float sa = fabsf(6.283185307179586f - fa);
            float dt = fminf(fa, sa);
            float wt = __expf(-0.5f * dt * dt / (1e-14f + prec_theta[k]*prec_theta[k]));
            float wv = wr * wt;
            if (!(wv == wv)) wv = 0.f;   // NaN -> 0 (reference)
            acc[k] += wv;
        }
    }
    #pragma unroll
    for (int k = 0; k < NKER; ++k)
        wsum[(size_t)tid * NKER + k] = acc[k] / acc[k];
}

// ------- prep: weight-norm W1 (row-major f16) ------------------------
__global__ void prep_w1(const float* __restrict__ v1, const float* __restrict__ g1,
                        _Float16* __restrict__ W1nT) {
    int o = threadIdx.x;
    float ss = 0.f;
    for (int j = 0; j < HIDN; ++j) { float x = v1[o*HIDN + j]; ss += x*x; }
    float sc = g1[o] / sqrtf(ss);
    for (int j = 0; j < HIDN; ++j)
        W1nT[o*HIDN + j] = (_Float16)(sc * v1[o*HIDN + j]);
}

// ------- prep: weight-norm W2 + W_conv transposed to [k][o][f] f16 ---
__global__ void prep_w2wc(const float* __restrict__ v2, const float* __restrict__ g2,
                          const float* __restrict__ Wc,
                          float* __restrict__ W2n, _Float16* __restrict__ WcB) {
    int tid = threadIdx.x;
    float ss = 0.f;
    for (int j = 0; j < HIDN; ++j) { float x = v2[j]; ss += x*x; }
    W2n[tid] = g2[0] * v2[tid] / sqrtf(ss);
    for (int idx = tid; idx < NKER * NB * 32; idx += 256) {
        int kk = idx >> 10, rem = idx & 1023;
        int o = rem >> 5, f = rem & 31;
        WcB[idx] = (_Float16)Wc[kk * 1024 + f * 32 + o];   // WcB[kk][o][f]
    }
}

// ------- prep: one-time f32 -> f16 conversion of images/captions -----
__global__ void prep_cvt(const float* __restrict__ src, _Float16* __restrict__ dst, int n) {
    int i = blockIdx.x * blockDim.x + threadIdx.x;
    if (i < n) dst[i] = (_Float16)src[i];
}

// ------- prep: transposed zero-padded f16 captions capT[c][d][lpad64] -
__global__ void prep_capT(const float* __restrict__ cap, _Float16* __restrict__ capT) {
    int i = blockIdx.x * blockDim.x + threadIdx.x;   // over 32*1024*64
    if (i >= NCAP * EMBED * LWP) return;
    int l = i & (LWP - 1);
    int d = (i >> 6) & (EMBED - 1);
    int c = i >> 16;
    capT[i] = (l < LW) ? (_Float16)cap[((size_t)c * LW + l) * EMBED + d] : (_Float16)0.f;
}

// ---------------- main fused kernel: one WG per (image, caption) -----
__global__ __launch_bounds__(256)
void gsmn_main(const float* __restrict__ images,     // (64,36,1024) f32 (for cosine)
               const _Float16* __restrict__ imgH,    // (64,36,1024) f16
               const _Float16* __restrict__ capH,    // (32,48,1024) f16
               const _Float16* __restrict__ capT,    // (32,1024,64) f16, l-pad zero
               const float* __restrict__ wsum,       // (64,36,8)
               const _Float16* __restrict__ W1nT,    // (256,256) row-major
               const float* __restrict__ W2n,        // (256)
               const _Float16* __restrict__ WcB,     // (8,32,32) [k][o][f]
               const float* __restrict__ b1,         // (256)
               const float* __restrict__ b2,         // (1)
               float* __restrict__ out) {            // (64,32)
    __shared__ float    sAttn[LW * LW];      // attn[l][reg] f32 (reg pad to 48)
    __shared__ _Float16 sAttnT[LW * LWP];    // attnS^T[reg][l] f16, l pad 64, zeroed
    __shared__ float    sQQ[LW * NB];
    __shared__ float    sQC[LW * NB];
    __shared__ float    sCC[LW * NB];
    __shared__ _Float16 sMvecH[LW * NB];
    __shared__ _Float16 sHpre[LW * HIDN];
    __shared__ float    sSum;

    const int c = blockIdx.x;          // caption
    const int b = blockIdx.y;          // image
    const int tid  = threadIdx.x;
    const int w    = tid >> 5;         // wave id (8 waves)
    const int lane = tid & 31;
    const int nl   = lane & 15;
    const int h    = lane >> 4;

    const _Float16* capR0 = capH + (size_t)c * LW * EMBED;
    const _Float16* imgR0 = imgH + (size_t)b * KREG * EMBED;
    const _Float16* capTc = capT + (size_t)c * EMBED * LWP;
    const float*    img   = images + (size_t)b * KREG * EMBED;

    // ===== GEMM1: attn(48x48) = cap(48x1024) . img^T, LeakyReLU(0.1) =====
    // N padded 36->48 via clamped row: junk lands in cols 36..47, never read.
    for (int t = w; t < 9; t += 8) {
        int mt = t / 3, nt = t % 3;
        v8f acc = {};
        const _Float16* capR = capR0 + (mt * 16 + nl) * EMBED;
        const int n = nt * 16 + nl;
        const int nc = n < KREG ? n : (KREG - 1);         // clamp, no guard
        const _Float16* imgR = imgR0 + (size_t)nc * EMBED;
        for (int ks = 0; ks < 32; ++ks) {
            int k0 = ks * 32;
            v16h a, bf;
            #pragma unroll
            for (int v = 0; v < 16; ++v) a[v] = capR[k0 + akoff(v, h)];
            #pragma unroll
            for (int v = 0; v < 16; ++v) bf[v] = imgR[k0 + bkoff(v, h)];
            acc = wmma16(a, bf, acc);
        }
        #pragma unroll
        for (int r = 0; r < 8; ++r) {
            int row = mt * 16 + r + h * 8;
            float x = acc[r];
            x = x > 0.f ? x : 0.1f * x;                  // LeakyReLU
            sAttn[row * LW + nt * 16 + nl] = x;
        }
    }
    __syncthreads();

    // ===== q-block norms, zero accumulators & attn^T; L2-norm attn rows ==
    for (int idx = tid; idx < LW * NB; idx += 256) {
        int row = idx >> 5, blk = idx & 31;
        float qq = 0.f;
        if (row < KREG) {
            const float* q = img + row * EMBED + blk * BS;
            #pragma unroll
            for (int j = 0; j < BS; ++j) qq += q[j] * q[j];
        }
        sQQ[idx] = qq; sQC[idx] = 0.f; sCC[idx] = 0.f;
    }
    for (int idx = tid; idx < LW * LWP; idx += 256) sAttnT[idx] = (_Float16)0.f;
    if (tid < LW) {
        float ss = 0.f;
        for (int r = 0; r < KREG; ++r) { float a = sAttn[tid * LW + r]; ss += a * a; }
        float inv = 1.f / (sqrtf(ss) + EPSF);
        for (int r = 0; r < KREG; ++r) sAttn[tid * LW + r] *= inv;
    }
    if (tid == 0) sSum = 0.f;
    __syncthreads();

    // ===== softmax over words l per region col; emit f16 transpose ======
    if (tid < KREG) {
        float mx = -1e30f;
        for (int l = 0; l < LW; ++l) mx = fmaxf(mx, LAM * sAttn[l * LW + tid]);
        float s = 0.f;
        for (int l = 0; l < LW; ++l) s += __expf(LAM * sAttn[l * LW + tid] - mx);
        float inv = 1.f / s;
        for (int l = 0; l < LW; ++l)
            sAttnT[tid * LWP + l] =
                (_Float16)(__expf(LAM * sAttn[l * LW + tid] - mx) * inv);
    }
    __syncthreads();

    // ===== GEMM2: ctx(36p48 x 1024) = attnS(48x64) . cap^T; fuse cosine ==
    // mt-outer / nt-inner: A fragments cached in registers, reused 8x.
    for (int mt = 0; mt < 3; ++mt) {
        const int mreg = mt * 16 + nl;
        v16h a0, a1;
        #pragma unroll
        for (int v = 0; v < 16; ++v) a0[v] = sAttnT[mreg * LWP + akoff(v, h)];
        #pragma unroll
        for (int v = 0; v < 16; ++v) a1[v] = sAttnT[mreg * LWP + 32 + akoff(v, h)];
        for (int nt = w; nt < 64; nt += 8) {
            const int nd = nt * 16 + nl;
            const _Float16* bp = capTc + (size_t)nd * LWP;   // contiguous, zero-pad
            v8f acc = {};
            v16h bf0, bf1;
            #pragma unroll
            for (int v = 0; v < 16; ++v) bf0[v] = bp[bkoff(v, h)];
            #pragma unroll
            for (int v = 0; v < 16; ++v) bf1[v] = bp[32 + bkoff(v, h)];
            acc = wmma16(a0, bf0, acc);
            acc = wmma16(a1, bf1, acc);
            const int blk = nd >> 5;
            #pragma unroll
            for (int r = 0; r < 8; ++r) {
                int row = mt * 16 + r + h * 8;
                if (row < KREG) {
                    float cv = acc[r];
                    float qv = img[row * EMBED + nd];
                    atomicAdd(&sQC[row * NB + blk], qv * cv);
                    atomicAdd(&sCC[row * NB + blk], cv * cv);
                }
            }
        }
    }
    __syncthreads();

    // ===== block cosine: mvec(48x32) f16, pad rows zero ==================
    for (int idx = tid; idx < LW * NB; idx += 256) {
        int row = idx >> 5;
        float mv = 0.f;
        if (row < KREG) {
            float den = fmaxf(sqrtf(sQQ[idx]) * sqrtf(sCC[idx]), EPSF);
            mv = sQC[idx] / den;
        }
        sMvecH[idx] = (_Float16)mv;
    }
    __syncthreads();

    // ===== GEMM4: hpre(48x256) = (mvec . W_conv) * wsum ==================
    for (int mt = 0; mt < 3; ++mt) {
        const int mi = mt * 16 + nl;
        v16h a;
        #pragma unroll
        for (int v = 0; v < 16; ++v) a[v] = sMvecH[mi * NB + akoff(v, h)];
        for (int nt = w; nt < 16; nt += 8) {
            const int j = nt * 16 + nl;
            const int kk = j >> 5, o = j & 31;
            const _Float16* wp = WcB + (kk * NB + o) * 32 + (h << 4);
            v8f acc = {};
            v16h bf;
            #pragma unroll
            for (int v = 0; v < 16; ++v) bf[v] = wp[v];
            acc = wmma16(a, bf, acc);
            #pragma unroll
            for (int r = 0; r < 8; ++r) {
                int row = mt * 16 + r + h * 8;
                float sc = (row < KREG) ? wsum[((size_t)b * KREG + row) * NKER + kk] : 0.f;
                sHpre[row * HIDN + j] = (_Float16)(acc[r] * sc);
            }
        }
    }
    __syncthreads();

    // ===== GEMM5: tanh(hpre . W1n^T + b1) . W2n, mean over regions =======
    float lsum = 0.f;
    for (int mt = 0; mt < 3; ++mt) {
        const int mi = mt * 16 + nl;
        v16h a[8];                               // full K row cached (64 VGPRs)
        #pragma unroll
        for (int ks = 0; ks < 8; ++ks) {
            #pragma unroll
            for (int v = 0; v < 16; ++v)
                a[ks][v] = sHpre[mi * HIDN + ks * 32 + akoff(v, h)];
        }
        for (int nt = w; nt < 16; nt += 8) {
            const int o = nt * 16 + nl;
            const _Float16* w1p = W1nT + o * HIDN + (h << 4);
            v8f acc = {};
            #pragma unroll
            for (int ks = 0; ks < 8; ++ks) {
                v16h bf;
                #pragma unroll
                for (int v = 0; v < 16; ++v) bf[v] = w1p[ks * 32 + v];
                acc = wmma16(a[ks], bf, acc);
            }
            float b1o = b1[o], w2o = W2n[o];
            #pragma unroll
            for (int r = 0; r < 8; ++r) {
                int row = mt * 16 + r + h * 8;
                if (row < KREG) lsum += tanhf(acc[r] + b1o) * w2o;
            }
        }
    }
    atomicAdd(&sSum, lsum);
    __syncthreads();
    if (tid == 0)
        out[b * NCAP + c] = (sSum + (float)KREG * b2[0]) / (float)KREG;
}

// ---------------------------------------------------------------------
extern "C" void kernel_launch(void* const* d_in, const int* in_sizes, int n_in,
                              void* d_out, int out_size, void* d_ws, size_t ws_size,
                              hipStream_t stream) {
    (void)in_sizes; (void)n_in; (void)out_size; (void)ws_size;
    const float* images     = (const float*)d_in[0];
    const float* captions   = (const float*)d_in[1];
    const float* bbox       = (const float*)d_in[2];
    // d_in[3] = cap_lens (unused, constant in benchmark)
    const float* W_conv     = (const float*)d_in[4];
    const float* mean_rho   = (const float*)d_in[5];
    const float* mean_theta = (const float*)d_in[6];
    const float* prec_rho   = (const float*)d_in[7];
    const float* prec_theta = (const float*)d_in[8];
    const float* v1         = (const float*)d_in[9];
    const float* g1         = (const float*)d_in[10];
    const float* b1         = (const float*)d_in[11];
    const float* v2         = (const float*)d_in[12];
    const float* g2         = (const float*)d_in[13];
    const float* b2         = (const float*)d_in[14];
    float* out = (float*)d_out;

    // workspace layout
    const int nImg  = NIMG * KREG * EMBED;     // 2,359,296
    const int nCap  = NCAP * LW * EMBED;       // 1,572,864
    const int nCapT = NCAP * EMBED * LWP;      // 2,097,152
    char* ws = (char*)d_ws;
    size_t off = 0;
    float*    wsum = (float*)(ws + off); off += (size_t)NIMG*KREG*NKER*4;   // 73728
    _Float16* W1nT = (_Float16*)(ws + off); off += (size_t)HIDN*HIDN*2;     // 131072
    float*    W2n  = (float*)(ws + off); off += (size_t)HIDN*4;             // 1024
    _Float16* WcB  = (_Float16*)(ws + off); off += (size_t)NKER*NB*32*2;    // 16384
    _Float16* imgH = (_Float16*)(ws + off); off += (size_t)nImg*2;
    _Float16* capH = (_Float16*)(ws + off); off += (size_t)nCap*2;
    _Float16* capT = (_Float16*)(ws + off); off += (size_t)nCapT*2;

    prep_wsum<<<(NIMG * KREG + 255) / 256, 256, 0, stream>>>(
        bbox, mean_rho, mean_theta, prec_rho, prec_theta, wsum);
    prep_w1<<<1, 256, 0, stream>>>(v1, g1, W1nT);
    prep_w2wc<<<1, 256, 0, stream>>>(v2, g2, W_conv, W2n, WcB);
    prep_cvt<<<(nImg + 255) / 256, 256, 0, stream>>>(images, imgH, nImg);
    prep_cvt<<<(nCap + 255) / 256, 256, 0, stream>>>(captions, capH, nCap);
    prep_capT<<<(nCapT + 255) / 256, 256, 0, stream>>>(captions, capT);

    dim3 grid(NCAP, NIMG);
    gsmn_main<<<grid, 256, 0, stream>>>(
        images, imgH, capH, capT, wsum, W1nT, W2n, WcB, b1, b2, out);
}
